// Attention_69999376990213
// MI455X (gfx1250) — matrix operand order repaired and checked by reading, not compiled, not run on hardware
//
#include <hip/hip_runtime.h>

typedef __bf16 bf16_t;
typedef __attribute__((ext_vector_type(16))) __bf16 bf16x16;
typedef __attribute__((ext_vector_type(8)))  float   f32x8;
typedef __attribute__((ext_vector_type(4)))  int     i32x4;

#define S_LEN 2048
#define H_DIM 2048
#define NH    16
#define NKV   4
#define HD    128
#define HKV   512
#define NQKV  3072

union Frag16 {
  bf16x16      v;
  unsigned int u[8];
  uint4        q[2];
};

// ---------------------------------------------------------------------------
// gfx1250 async global->LDS copy (ASYNCcnt path), with sync fallback
// Builtin signature (from compiler diagnostic): (global int4*, local int4*,
// imm offset, imm cpol)
// ---------------------------------------------------------------------------
#if __has_builtin(__builtin_amdgcn_global_load_async_to_lds_b128)
#define HAVE_ASYNC_LDS 1
#endif

typedef __attribute__((address_space(1))) i32x4* as1_i32x4_p;
typedef __attribute__((address_space(3))) i32x4* as3_i32x4_p;

__device__ __forceinline__ void cp16_lds(const bf16_t* g, bf16_t* l) {
#if defined(HAVE_ASYNC_LDS)
  void* gp = (void*)g;
  void* lp = (void*)l;
  __builtin_amdgcn_global_load_async_to_lds_b128(
      (as1_i32x4_p)gp, (as3_i32x4_p)lp, 0, 0);
#else
  *reinterpret_cast<uint4*>(l) = *reinterpret_cast<const uint4*>(g);
#endif
}

__device__ __forceinline__ void wait_async_lds() {
#if defined(HAVE_ASYNC_LDS)
#if __has_builtin(__builtin_amdgcn_s_wait_asynccnt)
  __builtin_amdgcn_s_wait_asynccnt(0);
#else
  asm volatile("s_wait_asynccnt 0" ::: "memory");
#endif
#endif
}

// ---------------------------------------------------------------------------
// elementwise converts / weight packing (weights packed TRANSPOSED: [N][K])
// ---------------------------------------------------------------------------
__global__ void cvt_f32_to_bf16(const float* __restrict__ x, bf16_t* __restrict__ y, int n) {
  int i = blockIdx.x * blockDim.x + threadIdx.x;
  if (i < n) y[i] = (bf16_t)x[i];
}

// Wqkv_t[n][k], n in [0,3072): col n of Wq / Wk / Wv
__global__ void pack_wqkv_t_bf16(const float* __restrict__ Wq, const float* __restrict__ Wk,
                                 const float* __restrict__ Wv, bf16_t* __restrict__ Wt, int total) {
  int i = blockIdx.x * blockDim.x + threadIdx.x;
  if (i >= total) return;
  int n = i / H_DIM;     // output column (QKV space)
  int k = i % H_DIM;     // input feature
  float v;
  if (n < H_DIM)            v = Wq[(size_t)k * H_DIM + n];
  else if (n < H_DIM + HKV) v = Wk[(size_t)k * HKV + (n - H_DIM)];
  else                      v = Wv[(size_t)k * HKV + (n - H_DIM - HKV)];
  Wt[i] = (bf16_t)v;
}

// Wo_t[n][k] = Wo[k][n]
__global__ void transpose_f32_to_bf16(const float* __restrict__ src, bf16_t* __restrict__ dst,
                                      int total) {
  int i = blockIdx.x * blockDim.x + threadIdx.x;
  if (i >= total) return;
  int n = i / H_DIM;
  int k = i % H_DIM;
  dst[i] = (bf16_t)src[(size_t)k * H_DIM + n];
}

// ---------------------------------------------------------------------------
// bf16 GEMM: C[M,N] = A[M,K] @ Bt[N,K]^T, f32 accumulate.
// Block 128x256, 8 waves (wave tile 64x64 = 16 WMMA accs), K-step 32,
// double-buffered LDS filled by async global->LDS b128 copies.
// ---------------------------------------------------------------------------
#define BM 128
#define BN 256
#define BK 32
#define LPITCH 40   // 32 + 8 pad bf16 -> 80B rows (16B aligned, bank-friendly)

__global__ __launch_bounds__(256) void gemm_bf16_tn_f32(
    const bf16_t* __restrict__ A, const bf16_t* __restrict__ Bt, float* __restrict__ C,
    int M, int N, int K) {
  __shared__ __align__(16) bf16_t lA[2][BM * LPITCH];
  __shared__ __align__(16) bf16_t lB[2][BN * LPITCH];

  const int tid  = threadIdx.x;
  const int lane = tid & 31;
  const int wid  = tid >> 5;
  const int rowBlk = blockIdx.y * BM;
  const int colBlk = blockIdx.x * BN;
  const int rowW = (wid & 1) * 64;
  const int colW = (wid >> 1) * 64;
  const int hi = lane >> 4;
  const int lo = lane & 15;
  (void)M;

  f32x8 acc[4][4];
  #pragma unroll
  for (int i = 0; i < 4; i++)
    #pragma unroll
    for (int j = 0; j < 4; j++) acc[i][j] = {};

  auto issueTiles = [&](int kt, int buf) {
    // A tile: 128 rows x 32 k (contiguous k) -> 512 x 16B
    for (int v = tid; v < (BM * BK / 8); v += 256) {
      int m = v >> 2, kc = v & 3;
      cp16_lds(A + (size_t)(rowBlk + m) * K + kt + kc * 8,
               &lA[buf][m * LPITCH + kc * 8]);
    }
    // B tile: 256 n-rows x 32 k (contiguous k, already transposed in global)
    for (int v = tid; v < (BN * BK / 8); v += 256) {
      int n = v >> 2, kc = v & 3;
      cp16_lds(Bt + (size_t)(colBlk + n) * K + kt + kc * 8,
               &lB[buf][n * LPITCH + kc * 8]);
    }
  };

  const int nkt = K / BK;
  issueTiles(0, 0);
  wait_async_lds();
  __syncthreads();

  for (int it = 0; it < nkt; ++it) {
    const int buf = it & 1;
    if (it + 1 < nkt) issueTiles((it + 1) * BK, buf ^ 1);   // overlap with WMMA

    // A fragments (ISA 7.12.2 16-bit A 16x32: two contiguous 16B runs/lane)
    Frag16 af[4];
    #pragma unroll
    for (int mi = 0; mi < 4; mi++) {
      int r  = rowW + mi * 16 + lo;
      int kb = hi * 8;
      af[mi].q[0] = *reinterpret_cast<const uint4*>(&lA[buf][r * LPITCH + kb]);
      af[mi].q[1] = *reinterpret_cast<const uint4*>(&lA[buf][r * LPITCH + kb + 16]);
    }
    #pragma unroll
    for (int ni = 0; ni < 4; ni++) {
      Frag16 bf;
      int n  = colW + ni * 16 + lo;
      int kb = hi * 8;
      bf.q[0] = *reinterpret_cast<const uint4*>(&lB[buf][n * LPITCH + kb]);
      bf.q[1] = *reinterpret_cast<const uint4*>(&lB[buf][n * LPITCH + kb + 16]);
      #pragma unroll
      for (int mi = 0; mi < 4; mi++) {
        acc[mi][ni] = __builtin_amdgcn_wmma_f32_16x16x32_bf16(
            false, af[mi].v, false, bf.v, (short)0, acc[mi][ni], false, false);
      }
    }
    wait_async_lds();
    __syncthreads();
  }

  #pragma unroll
  for (int mi = 0; mi < 4; mi++) {
    #pragma unroll
    for (int ni = 0; ni < 4; ni++) {
      int col = colBlk + colW + ni * 16 + lo;
      #pragma unroll
      for (int r = 0; r < 8; r++) {
        int row = rowBlk + rowW + mi * 16 + r + 8 * hi;
        C[(size_t)row * N + col] = acc[mi][ni][r];
      }
    }
  }
}

// ---------------------------------------------------------------------------
// RoPE + pack. Q -> [S][16*128] bf16, K -> [S][4*128] bf16,
// V -> TRANSPOSED [kv][hd][S] bf16 so attention V tiles are straight copies.
// ---------------------------------------------------------------------------
__global__ __launch_bounds__(256) void rope_pack(
    const float* __restrict__ QKV, const int* __restrict__ pos,
    bf16_t* __restrict__ Qb, bf16_t* __restrict__ Kb, bf16_t* __restrict__ Vt) {
  int s = blockIdx.x;
  float p = (float)pos[s];
  const float* row = QKV + (size_t)s * NQKV;

  for (int i = threadIdx.x; i < H_DIM / 2; i += blockDim.x) {
    int di = i & 63;
    int c0 = (i >> 6) * HD + 2 * di;
    float inv = __powf(10000.0f, -(float)di * (1.0f / 64.0f));
    float a = p * inv, sn, cs;
    __sincosf(a, &sn, &cs);
    float x0 = row[c0], x1 = row[c0 + 1];
    Qb[(size_t)s * H_DIM + c0]     = (bf16_t)(x0 * cs - x1 * sn);
    Qb[(size_t)s * H_DIM + c0 + 1] = (bf16_t)(x0 * sn + x1 * cs);
  }
  for (int i = threadIdx.x; i < HKV / 2; i += blockDim.x) {
    int di = i & 63;
    int c0 = (i >> 6) * HD + 2 * di;
    float inv = __powf(10000.0f, -(float)di * (1.0f / 64.0f));
    float a = p * inv, sn, cs;
    __sincosf(a, &sn, &cs);
    float x0 = row[H_DIM + c0], x1 = row[H_DIM + c0 + 1];
    Kb[(size_t)s * HKV + c0]     = (bf16_t)(x0 * cs - x1 * sn);
    Kb[(size_t)s * HKV + c0 + 1] = (bf16_t)(x0 * sn + x1 * cs);
  }
  for (int i = threadIdx.x; i < HKV; i += blockDim.x) {
    // i = kv*128 + d ; Vt[(kv*HD + d)*S + s]
    Vt[(size_t)i * S_LEN + s] = (bf16_t)row[H_DIM + HKV + i];
  }
}

// ---------------------------------------------------------------------------
// Flash attention. grid = (S/128 q-blocks, 16 heads), 8 waves per block.
// Each wave: 16 queries, online softmax, key blocks of 32, double-buffered
// async K/V tiles.
// ---------------------------------------------------------------------------
#define KPITCH 136   // 128 + 8 pad bf16 -> 272B rows (16B aligned)
#define VPITCH 40    // 32 + 8 pad

__global__ __launch_bounds__(256) void flash_attn(
    const bf16_t* __restrict__ Qb, const bf16_t* __restrict__ Kb, const bf16_t* __restrict__ Vt,
    const int* __restrict__ amask, const int* __restrict__ seg,
    bf16_t* __restrict__ Ob) {
  __shared__ __align__(16) bf16_t lK[2][32 * KPITCH];   // [key][hd]
  __shared__ __align__(16) bf16_t lV[2][HD * VPITCH];   // [hd][key]
  __shared__ __align__(16) bf16_t lP[8][16 * 32];       // per-wave P staging

  const int tid  = threadIdx.x;
  const int lane = tid & 31;
  const int wid  = tid >> 5;
  const int hi   = lane >> 4;
  const int lo   = lane & 15;
  const int head = blockIdx.y;
  const int kv   = head >> 2;            // GQA: 4 query heads per kv head
  const int qblk = blockIdx.x;
  const int q0   = qblk * 128 + wid * 16;

  // Q fragments: 4 x (16x32) over hd=128, loaded straight in A layout (global)
  Frag16 qf[4];
  {
    const bf16_t* qrow = Qb + (size_t)(q0 + lo) * H_DIM + head * HD;
    #pragma unroll
    for (int f = 0; f < 4; f++) {
      int kb = f * 32 + hi * 8;
      qf[f].q[0] = *reinterpret_cast<const uint4*>(qrow + kb);
      qf[f].q[1] = *reinterpret_cast<const uint4*>(qrow + kb + 16);
    }
  }

  int segq[8];
  #pragma unroll
  for (int r = 0; r < 8; r++) segq[r] = seg[q0 + r + 8 * hi];

  float m[8], l[8];
  f32x8 accO[8];
  #pragma unroll
  for (int r = 0; r < 8; r++) { m[r] = -3.0e38f; l[r] = 0.0f; }
  #pragma unroll
  for (int c = 0; c < 8; c++) accO[c] = {};

  auto issueTiles = [&](int kstart, int buf) {
    // K tile: 32 keys x 128 hd, contiguous -> 512 x 16B
    for (int v = tid; v < (32 * HD / 8); v += 256) {
      int key = v >> 4, dc = v & 15;
      cp16_lds(Kb + (size_t)(kstart + key) * HKV + kv * HD + dc * 8,
               &lK[buf][key * KPITCH + dc * 8]);
    }
    // V tile from transposed global layout: 128 hd-rows x 32 keys contiguous
    for (int v = tid; v < (HD * 32 / 8); v += 256) {
      int d = v >> 2, kc = v & 3;
      cp16_lds(Vt + (size_t)(kv * HD + d) * S_LEN + kstart + kc * 8,
               &lV[buf][d * VPITCH + kc * 8]);
    }
  };

  const float scale = 0.08838834764831845f;   // 1/sqrt(128)
  const int nkb = (qblk + 1) * 4;             // causal

  issueTiles(0, 0);
  wait_async_lds();
  __syncthreads();

  for (int kb = 0; kb < nkb; kb++) {
    const int kstart = kb * 32;
    const int buf = kb & 1;
    if (kb + 1 < nkb) issueTiles(kstart + 32, buf ^ 1);   // overlap

    // S = Q K^T (two 16x16 key tiles; contract hd=128 in 4 WMMA each)
    f32x8 s0 = {}, s1 = {};
    #pragma unroll
    for (int f = 0; f < 4; f++) {
      Frag16 b0, b1;
      int kbase = f * 32 + hi * 8;
      b0.q[0] = *reinterpret_cast<const uint4*>(&lK[buf][lo * KPITCH + kbase]);
      b0.q[1] = *reinterpret_cast<const uint4*>(&lK[buf][lo * KPITCH + kbase + 16]);
      b1.q[0] = *reinterpret_cast<const uint4*>(&lK[buf][(lo + 16) * KPITCH + kbase]);
      b1.q[1] = *reinterpret_cast<const uint4*>(&lK[buf][(lo + 16) * KPITCH + kbase + 16]);
      s0 = __builtin_amdgcn_wmma_f32_16x16x32_bf16(false, qf[f].v, false, b0.v, (short)0, s0, false, false);
      s1 = __builtin_amdgcn_wmma_f32_16x16x32_bf16(false, qf[f].v, false, b1.v, (short)0, s1, false, false);
    }

    // masking + online softmax: row r+8*hi's 16 cols live in this 16-lane half
    const int  key0 = kstart + lo;
    const int  key1 = key0 + 16;
    const bool am0  = amask[key0] > 0;
    const bool am1  = amask[key1] > 0;
    const int  sg0  = seg[key0];
    const int  sg1  = seg[key1];
    float alpha[8];
    #pragma unroll
    for (int r = 0; r < 8; r++) {
      int qr = q0 + r + 8 * hi;
      float v0 = s0[r] * scale, v1 = s1[r] * scale;
      if (!(key0 <= qr && am0 && sg0 == segq[r])) v0 = -3.0e38f;
      if (!(key1 <= qr && am1 && sg1 == segq[r])) v1 = -3.0e38f;
      float rm = fmaxf(v0, v1);
      #pragma unroll
      for (int msk = 1; msk < 16; msk <<= 1) rm = fmaxf(rm, __shfl_xor(rm, msk, 32));
      float mn = fmaxf(m[r], rm);
      float al = __expf(m[r] - mn);
      float p0 = __expf(v0 - mn);
      float p1 = __expf(v1 - mn);
      float rs = p0 + p1;
      #pragma unroll
      for (int msk = 1; msk < 16; msk <<= 1) rs += __shfl_xor(rs, msk, 32);
      m[r] = mn;
      l[r] = l[r] * al + rs;
      alpha[r] = al;
      int prow = r + 8 * hi;
      lP[wid][prow * 32 + lo]      = (bf16_t)p0;
      lP[wid][prow * 32 + 16 + lo] = (bf16_t)p1;
    }
    #pragma unroll
    for (int c = 0; c < 8; c++)
      #pragma unroll
      for (int r = 0; r < 8; r++) accO[c][r] *= alpha[r];

    // same-wave LDS RAW on lP (layout change): DS in-order per wave, wait data
    asm volatile("s_wait_dscnt 0" ::: "memory");

    // P fragment (16x32, A layout): two contiguous 16B runs
    Frag16 pf;
    {
      int kb2 = hi * 8;
      pf.q[0] = *reinterpret_cast<const uint4*>(&lP[wid][lo * 32 + kb2]);
      pf.q[1] = *reinterpret_cast<const uint4*>(&lP[wid][lo * 32 + kb2 + 16]);
    }
    // O += P @ V  (8 x 16-col chunks of hd)
    #pragma unroll
    for (int c = 0; c < 8; c++) {
      Frag16 vf;
      int n = c * 16 + lo;
      vf.q[0] = *reinterpret_cast<const uint4*>(&lV[buf][n * VPITCH + hi * 8]);
      vf.q[1] = *reinterpret_cast<const uint4*>(&lV[buf][n * VPITCH + hi * 8 + 16]);
      accO[c] = __builtin_amdgcn_wmma_f32_16x16x32_bf16(
          false, pf.v, false, vf.v, (short)0, accO[c], false, false);
    }
    wait_async_lds();
    __syncthreads();
  }

  #pragma unroll
  for (int c = 0; c < 8; c++) {
    int col = head * HD + c * 16 + lo;
    #pragma unroll
    for (int r = 0; r < 8; r++) {
      int row = q0 + r + 8 * hi;
      Ob[(size_t)row * H_DIM + col] = (bf16_t)(accO[c][r] / l[r]);
    }
  }
}

// ---------------------------------------------------------------------------
// host launcher
// ---------------------------------------------------------------------------
extern "C" void kernel_launch(void* const* d_in, const int* in_sizes, int n_in,
                              void* d_out, int out_size, void* d_ws, size_t ws_size,
                              hipStream_t stream) {
  const float* hs    = (const float*)d_in[0];
  const int*   amask = (const int*)d_in[1];
  const int*   seg   = (const int*)d_in[2];
  const int*   pos   = (const int*)d_in[3];
  const float* Wq    = (const float*)d_in[4];
  const float* Wk    = (const float*)d_in[5];
  const float* Wv    = (const float*)d_in[6];
  const float* Wo    = (const float*)d_in[7];
  float* out = (float*)d_out;

  char* ws = (char*)d_ws;
  size_t off = 0;
  auto alloc = [&](size_t bytes) -> void* {
    void* p = (void*)(ws + off);
    off += (bytes + 255) & ~(size_t)255;
    return p;
  };

  bf16_t* Xb     = (bf16_t*)alloc((size_t)S_LEN * H_DIM * 2);  // hidden bf16
  bf16_t* Wqkvt  = (bf16_t*)alloc((size_t)NQKV * H_DIM * 2);   // fused W, [N][K]
  bf16_t* Wot    = (bf16_t*)alloc((size_t)H_DIM * H_DIM * 2);  // Wo^T, [N][K]
  float*  QKV    = (float*)alloc((size_t)S_LEN * NQKV * 4);    // projection out
  bf16_t* Qb     = (bf16_t*)alloc((size_t)S_LEN * H_DIM * 2);  // roped Q
  bf16_t* Kb     = (bf16_t*)alloc((size_t)S_LEN * HKV * 2);    // roped K
  bf16_t* Vt     = (bf16_t*)alloc((size_t)NKV * HD * S_LEN * 2); // V [kv][hd][S]
  bf16_t* Ab     = (bf16_t*)alloc((size_t)S_LEN * H_DIM * 2);  // attention out
  (void)ws_size; (void)n_in; (void)in_sizes; (void)out_size;

  {
    int n = S_LEN * H_DIM;
    cvt_f32_to_bf16<<<(n + 255) / 256, 256, 0, stream>>>(hs, Xb, n);
    transpose_f32_to_bf16<<<(n + 255) / 256, 256, 0, stream>>>(Wo, Wot, n);
    int t = NQKV * H_DIM;
    pack_wqkv_t_bf16<<<(t + 255) / 256, 256, 0, stream>>>(Wq, Wk, Wv, Wqkvt, t);
  }

  // fused QKV projection
  gemm_bf16_tn_f32<<<dim3(NQKV / BN, S_LEN / BM), 256, 0, stream>>>(
      Xb, Wqkvt, QKV, S_LEN, NQKV, H_DIM);

  // RoPE + pack
  rope_pack<<<S_LEN, 256, 0, stream>>>(QKV, pos, Qb, Kb, Vt);

  // flash attention
  flash_attn<<<dim3(S_LEN / 128, NH), 256, 0, stream>>>(Qb, Kb, Vt, amask, seg, Ab);

  // output projection
  gemm_bf16_tn_f32<<<dim3(H_DIM / BN, S_LEN / BM), 256, 0, stream>>>(
      Ab, Wot, out, S_LEN, H_DIM, H_DIM);
}
